// MultiHeadAttentionV2_79697413145007
// MI455X (gfx1250) — matrix operand order repaired and checked by reading, not compiled, not run on hardware
//
#include <hip/hip_runtime.h>
#include <math.h>

// ---------------------------------------------------------------------------
// MultiHeadAttentionV2 forward for gfx1250 (CDNA5), bf16 WMMA pipeline.
//  k1: qkv = x@Wqkv+b -> bf16 Q/K/V [B,H,L,hd]
//  k2: fused scores + rel-bias + mask + softmax + attn@V (flash, one pass)
//  k3: out = ctx@Wout+bout (WMMA)
// ---------------------------------------------------------------------------

typedef __attribute__((ext_vector_type(16))) __bf16 v16bf;
typedef __attribute__((ext_vector_type(2)))  __bf16 v2bf;
typedef __attribute__((ext_vector_type(8)))  float  v8f;

#define B_SZ   2
#define L_SZ   2048
#define D_SZ   1024
#define H_SZ   16
#define HD_SZ  64
#define N3_SZ  3072
#define MAXREL 128

static __device__ __forceinline__ v8f wmma_bf16(v16bf a, v16bf b, v8f c) {
  return __builtin_amdgcn_wmma_f32_16x16x32_bf16(
      false, a, false, b, (short)0, c, false, false);
}

static __device__ __forceinline__ void store_pair_f32(__bf16* dst, float a, float b) {
  v2bf p; p[0] = (__bf16)a; p[1] = (__bf16)b;
  *(v2bf*)dst = p;
}

// A fragment from LDS: lane holds row (lane&15); half selects K-groups.
static __device__ __forceinline__ v16bf load_a_frag(const __bf16* rowPtr, int half) {
  v16bf a;
#pragma unroll
  for (int i = 0; i < 8; ++i) a[i] = rowPtr[half * 16 + i];
#pragma unroll
  for (int i = 0; i < 8; ++i) a[8 + i] = rowPtr[half * 16 + 8 + i];
  return a;
}
// B fragment from [N][K]-transposed LDS tile: lane holds column (lane&15).
static __device__ __forceinline__ v16bf load_b_frag(const __bf16* rowPtr, int half) {
  v16bf b;
#pragma unroll
  for (int i = 0; i < 16; ++i) b[i] = rowPtr[half * 16 + i];
  return b;
}
// Fragment built from two 16B global loads (global_load_b128 x2).
static __device__ __forceinline__ v16bf frag_2x16B(const __bf16* p0, const __bf16* p1) {
  union { uint4 u[2]; v16bf v; } t;
  t.u[0] = *(const uint4*)p0;
  t.u[1] = *(const uint4*)p1;
  return t.v;
}

// ---------------------------------------------------------------------------
// Kernel 1: qkv = x @ Wqkv + bqkv ; scatter into bf16 Q/K/V [B,H,L,hd]
// Block tile 64(M) x 128(N); 8 waves as 2x4; wave tile 32x32 (4 wmma/K-step).
// ---------------------------------------------------------------------------
__global__ __launch_bounds__(256) void qkv_gemm(
    const float* __restrict__ x, const float* __restrict__ Wqkv,
    const float* __restrict__ bqkv,
    __bf16* __restrict__ qbf, __bf16* __restrict__ kbf, __bf16* __restrict__ vbf) {
  __shared__ __bf16 As[64 * 40];    // [m][k], stride 40 elems (80B)
  __shared__ __bf16 Bs[128 * 40];   // [n][k] transposed
  const int tid = threadIdx.x;
  const int lane = tid & 31, wave = tid >> 5;
  const int half = lane >> 4, row16 = lane & 15;
  const int wr = wave >> 2, wc = wave & 3;
  const int m0 = blockIdx.y * 64;
  const int n0 = blockIdx.x * 128;

  v8f acc00 = {}, acc01 = {}, acc10 = {}, acc11 = {};
  for (int k0 = 0; k0 < D_SZ; k0 += 32) {
#pragma unroll
    for (int t = 0; t < 4; ++t) {             // A: 64x32 fp32, 1024 pairs
      int idx = tid + t * 256;
      int r = idx >> 4, c2 = (idx & 15) * 2;
      const float* s = &x[(size_t)(m0 + r) * D_SZ + k0 + c2];
      store_pair_f32(&As[r * 40 + c2], s[0], s[1]);
    }
#pragma unroll
    for (int t = 0; t < 8; ++t) {             // B: 32(K)x128(N) -> Bs[n][k]
      int idx = tid + t * 256;
      int n = idx & 127, k2 = (idx >> 7) * 2;
      store_pair_f32(&Bs[n * 40 + k2],
                     Wqkv[(size_t)(k0 + k2) * N3_SZ + n0 + n],
                     Wqkv[(size_t)(k0 + k2 + 1) * N3_SZ + n0 + n]);
    }
    if (k0 + 32 < D_SZ) {
      __builtin_prefetch(&x[(size_t)(m0 + (tid >> 2)) * D_SZ + k0 + 32], 0, 3);
      __builtin_prefetch(&Wqkv[(size_t)(k0 + 32 + (tid >> 7)) * N3_SZ + n0 + (tid & 127)], 0, 3);
    }
    __syncthreads();
    v16bf a0 = load_a_frag(&As[(wr * 32 + row16) * 40], half);
    v16bf a1 = load_a_frag(&As[(wr * 32 + 16 + row16) * 40], half);
    v16bf b0 = load_b_frag(&Bs[(wc * 32 + row16) * 40], half);
    v16bf b1 = load_b_frag(&Bs[(wc * 32 + 16 + row16) * 40], half);
    acc00 = wmma_bf16(a0, b0, acc00);
    acc01 = wmma_bf16(a0, b1, acc01);
    acc10 = wmma_bf16(a1, b0, acc10);
    acc11 = wmma_bf16(a1, b1, acc11);
    __syncthreads();
  }

#pragma unroll
  for (int mt = 0; mt < 2; ++mt) {
#pragma unroll
    for (int nt = 0; nt < 2; ++nt) {
      v8f acc = mt == 0 ? (nt == 0 ? acc00 : acc01) : (nt == 0 ? acc10 : acc11);
#pragma unroll
      for (int r = 0; r < 8; ++r) {
        int m = m0 + wr * 32 + mt * 16 + r + 8 * half;
        int n = n0 + wc * 32 + nt * 16 + row16;
        float v = acc[r] + bqkv[n];
        int which = n >> 10;                  // 0=q 1=k 2=v
        int d = n & (D_SZ - 1);
        int h = d >> 6, di = d & 63;
        int b = m >> 11, l = m & (L_SZ - 1);
        size_t dst = ((size_t)(b * H_SZ + h) * L_SZ + l) * HD_SZ + di;
        __bf16 bv = (__bf16)v;
        if (which == 0)      qbf[dst] = bv;
        else if (which == 1) kbf[dst] = bv;
        else                 vbf[dst] = bv;
      }
    }
  }
}

// ---------------------------------------------------------------------------
// Kernel 2 (flash-fused): attn = softmax(scale*QK^T + rel_bias, masked),
// written to d_out, AND ctx = attn @ V accumulated in the same kernel.
// One block = 16 q-rows x full 2048 keys for one (b,h). 8 waves; wave w owns
// key-tiles j*8+w (j=0..15) with 16 resident v8f prob accumulators.
// Phase 2 transposes prob tiles (C-layout -> A-layout) through wave-private
// LDS (hardware-ordered, no barriers), stages V^T likewise, and reduces the
// per-wave 16x64 ctx partials across waves with ds_add_f32.
// ---------------------------------------------------------------------------
__global__ __launch_bounds__(256) void attn_flash_kernel(
    const __bf16* __restrict__ qbf, const __bf16* __restrict__ kbf,
    const __bf16* __restrict__ vbf,
    const float* __restrict__ rel_bias, const int* __restrict__ mask,
    float* __restrict__ attn, __bf16* __restrict__ ctxbf) {
  __shared__ __bf16 Pls[8][16 * 40];   // per-wave prob pair, [q][K0..31]
  __shared__ __bf16 Vls[8][64 * 40];   // per-wave V^T pair, [hd][key0..31]
  __shared__ float  ctxred[16 * 64];   // cross-wave ctx accumulator
  __shared__ float  red[8][16];        // softmax cross-wave reduce

  const int tid = threadIdx.x;
  const int lane = tid & 31, w = tid >> 5;
  const int half = lane >> 4, row16 = lane & 15;
  const int bh = blockIdx.y;
  const int b = bh >> 4, h = bh & 15;
  const int q0 = blockIdx.x * 16;
  const __bf16* Qg = qbf + (size_t)bh * L_SZ * HD_SZ;
  const __bf16* Kg = kbf + (size_t)bh * L_SZ * HD_SZ;
  const __bf16* Vg = vbf + (size_t)bh * L_SZ * HD_SZ;
  const float* relh = rel_bias + h * (2 * MAXREL - 1);
  const int* maskb = mask + b * L_SZ;

  // Q fragments for kk = 0 and 32, loaded once from global.
  const __bf16* qrow = Qg + (size_t)(q0 + row16) * HD_SZ;
  v16bf aq0 = frag_2x16B(qrow + half * 16,      qrow + half * 16 + 8);
  v16bf aq1 = frag_2x16B(qrow + 32 + half * 16, qrow + 40 + half * 16);

  v8f acc[16];
#pragma unroll
  for (int j = 0; j < 16; ++j) acc[j] = (v8f){};

  // ---- Phase 1a: scores = scale*QK^T + bias + mask (32 wmma / wave)
  const float scale = 0.125f;                 // 1/sqrt(64)
#pragma unroll
  for (int j = 0; j < 16; ++j) {
    const int kb = (j * 8 + w) * 16;          // this wave's key-tile base
    const __bf16* krow = Kg + (size_t)(kb + row16) * HD_SZ + half * 16;
    v16bf b0 = frag_2x16B(krow,      krow + 8);       // kk = 0
    v16bf b1 = frag_2x16B(krow + 32, krow + 40);      // kk = 32
    acc[j] = wmma_bf16(aq0, b0, acc[j]);
    acc[j] = wmma_bf16(aq1, b1, acc[j]);
    const int key = kb + row16;
    const float mval = (maskb[key] == 0) ? -INFINITY : 0.0f;
#pragma unroll
    for (int r = 0; r < 8; ++r) {
      int q = q0 + r + 8 * half;
      int d = q - key;
      d = d < -(MAXREL - 1) ? -(MAXREL - 1) : (d > (MAXREL - 1) ? (MAXREL - 1) : d);
      acc[j][r] = acc[j][r] * scale + relh[d + (MAXREL - 1)] + mval;
    }
  }

  // ---- Phase 1b: softmax (shuffle within wave halves, LDS across waves)
  v8f pm = acc[0];
#pragma unroll
  for (int j = 1; j < 16; ++j)
#pragma unroll
    for (int r = 0; r < 8; ++r) pm[r] = fmaxf(pm[r], acc[j][r]);
#pragma unroll
  for (int off = 1; off <= 8; off <<= 1)
#pragma unroll
    for (int r = 0; r < 8; ++r) pm[r] = fmaxf(pm[r], __shfl_xor(pm[r], off, 32));
  if (row16 == 0) {
#pragma unroll
    for (int r = 0; r < 8; ++r) red[w][r + 8 * half] = pm[r];
  }
  __syncthreads();
  float rowmax[8];
#pragma unroll
  for (int r = 0; r < 8; ++r) {
    float m = red[0][r + 8 * half];
#pragma unroll
    for (int w2 = 1; w2 < 8; ++w2) m = fmaxf(m, red[w2][r + 8 * half]);
    rowmax[r] = m;
  }
  __syncthreads();

  v8f ps = {};
#pragma unroll
  for (int j = 0; j < 16; ++j)
#pragma unroll
    for (int r = 0; r < 8; ++r) {
      float e = __expf(acc[j][r] - rowmax[r]);
      acc[j][r] = e;
      ps[r] += e;
    }
#pragma unroll
  for (int off = 1; off <= 8; off <<= 1)
#pragma unroll
    for (int r = 0; r < 8; ++r) ps[r] += __shfl_xor(ps[r], off, 32);
  if (row16 == 0) {
#pragma unroll
    for (int r = 0; r < 8; ++r) red[w][r + 8 * half] = ps[r];
  }
  __syncthreads();
  float inv[8];
#pragma unroll
  for (int r = 0; r < 8; ++r) {
    float s = red[0][r + 8 * half];
#pragma unroll
    for (int w2 = 1; w2 < 8; ++w2) s += red[w2][r + 8 * half];
    inv[r] = 1.0f / s;
  }

  // zero cross-wave ctx accumulator (barrier below orders it vs ds_add)
#pragma unroll
  for (int t = 0; t < 4; ++t) ctxred[tid + t * 256] = 0.0f;

  // ---- Phase 1c: normalize in-register + write attn (mandatory traffic)
  float* arow = attn + ((size_t)bh * L_SZ + q0) * L_SZ;
#pragma unroll
  for (int j = 0; j < 16; ++j) {
    const int key = (j * 8 + w) * 16 + row16;
#pragma unroll
    for (int r = 0; r < 8; ++r) {
      float p = acc[j][r] * inv[r];
      acc[j][r] = p;
      arow[(size_t)(r + 8 * half) * L_SZ + key] = p;
    }
  }
  __syncthreads();

  // ---- Phase 2: ctx partial = probs(16x256) @ V(256x64), per wave.
  // Process prob tiles in pairs (K=32). Wave-private LDS, no barriers.
  __bf16* P  = &Pls[w][0];
  __bf16* Vt = &Vls[w][0];
  v8f c0 = {}, c1 = {}, c2 = {}, c3 = {};
#pragma unroll
  for (int jj = 0; jj < 8; ++jj) {
    const int kbA = ((2 * jj)     * 8 + w) * 16;
    const int kbB = ((2 * jj + 1) * 8 + w) * 16;
    // stage prob pair: C-layout -> A-layout [q][K0..31] (bf16)
#pragma unroll
    for (int t = 0; t < 2; ++t)
#pragma unroll
      for (int r = 0; r < 8; ++r)
        P[(r + 8 * half) * 40 + t * 16 + row16] = (__bf16)acc[2 * jj + t][r];
    // stage V^T pair: lane c=0..31 handles one key (c<16 -> kbA, else kbB)
    {
      const int c = lane;
      const int keyg = (c < 16) ? (kbA + c) : (kbB + (c - 16));
      const __bf16* vrow = Vg + (size_t)keyg * HD_SZ;
#pragma unroll
      for (int hd = 0; hd < HD_SZ; ++hd) Vt[hd * 40 + c] = vrow[hd];
    }
    // 4 wmma: ctx[16 x 64] += P[16x32] x V[32x64]
    v16bf af = load_a_frag(&P[row16 * 40], half);
    v16bf b0 = load_b_frag(&Vt[(0  + row16) * 40], half);
    v16bf b1 = load_b_frag(&Vt[(16 + row16) * 40], half);
    v16bf b2 = load_b_frag(&Vt[(32 + row16) * 40], half);
    v16bf b3 = load_b_frag(&Vt[(48 + row16) * 40], half);
    c0 = wmma_bf16(af, b0, c0);
    c1 = wmma_bf16(af, b1, c1);
    c2 = wmma_bf16(af, b2, c2);
    c3 = wmma_bf16(af, b3, c3);
  }

  // cross-wave reduction with LDS float atomics (ds_add_f32)
#pragma unroll
  for (int nt = 0; nt < 4; ++nt) {
    v8f c = nt == 0 ? c0 : (nt == 1 ? c1 : (nt == 2 ? c2 : c3));
#pragma unroll
    for (int r = 0; r < 8; ++r) {
      int q = r + 8 * half;
      int n = nt * 16 + row16;
      atomicAdd(&ctxred[q * 64 + n], c[r]);
    }
  }
  __syncthreads();

  // write ctx as bf16 into [B,L,D] layout (D = h*64 + n)
#pragma unroll
  for (int t = 0; t < 4; ++t) {
    int idx = tid + t * 256;
    int q = idx >> 6, n = idx & 63;
    ctxbf[((size_t)(b * L_SZ + q0 + q)) * D_SZ + h * HD_SZ + n] =
        (__bf16)ctxred[idx];
  }
}

// ---------------------------------------------------------------------------
// Kernel 3: out = ctx @ Wout + bout (fp32 result to d_out)
// Block tile 64(M) x 128(N); wave tile 32x32 (4 wmma/K-step).
// ---------------------------------------------------------------------------
__global__ __launch_bounds__(256) void out_gemm(
    const __bf16* __restrict__ ctxbf, const float* __restrict__ Wout,
    const float* __restrict__ bout, float* __restrict__ out) {
  __shared__ __bf16 As[64 * 40];
  __shared__ __bf16 Bs[128 * 40];
  const int tid = threadIdx.x;
  const int lane = tid & 31, wave = tid >> 5;
  const int half = lane >> 4, row16 = lane & 15;
  const int wr = wave >> 2, wc = wave & 3;
  const int m0 = blockIdx.y * 64;
  const int n0 = blockIdx.x * 128;
  const __bf16* Ag = ctxbf + (size_t)m0 * D_SZ;

  v8f acc00 = {}, acc01 = {}, acc10 = {}, acc11 = {};
  for (int k0 = 0; k0 < D_SZ; k0 += 32) {
    {                                         // A: 64x32 bf16, 16B chunks
      int r = tid >> 2, c = tid & 3;
      *(uint4*)&As[r * 40 + c * 8] =
          *(const uint4*)&Ag[(size_t)r * D_SZ + k0 + c * 8];
    }
#pragma unroll
    for (int t = 0; t < 8; ++t) {             // B: 32(K)x128(N) -> Bs[n][k]
      int idx = tid + t * 256;
      int n = idx & 127, k2 = (idx >> 7) * 2;
      store_pair_f32(&Bs[n * 40 + k2],
                     Wout[(size_t)(k0 + k2) * D_SZ + n0 + n],
                     Wout[(size_t)(k0 + k2 + 1) * D_SZ + n0 + n]);
    }
    if (k0 + 32 < D_SZ)
      __builtin_prefetch(&Wout[(size_t)(k0 + 32 + (tid >> 7)) * D_SZ + n0 + (tid & 127)], 0, 3);
    __syncthreads();
    v16bf a0 = load_a_frag(&As[(wr * 32 + row16) * 40], half);
    v16bf a1 = load_a_frag(&As[(wr * 32 + 16 + row16) * 40], half);
    v16bf b0 = load_b_frag(&Bs[(wc * 32 + row16) * 40], half);
    v16bf b1 = load_b_frag(&Bs[(wc * 32 + 16 + row16) * 40], half);
    acc00 = wmma_bf16(a0, b0, acc00);
    acc01 = wmma_bf16(a0, b1, acc01);
    acc10 = wmma_bf16(a1, b0, acc10);
    acc11 = wmma_bf16(a1, b1, acc11);
    __syncthreads();
  }

#pragma unroll
  for (int mt = 0; mt < 2; ++mt) {
#pragma unroll
    for (int nt = 0; nt < 2; ++nt) {
      v8f acc = mt == 0 ? (nt == 0 ? acc00 : acc01) : (nt == 0 ? acc10 : acc11);
#pragma unroll
      for (int r = 0; r < 8; ++r) {
        int m = m0 + wr * 32 + mt * 16 + r + 8 * half;
        int n = n0 + wc * 32 + nt * 16 + row16;
        out[(size_t)m * D_SZ + n] = acc[r] + bout[n];
      }
    }
  }
}

// ---------------------------------------------------------------------------
extern "C" void kernel_launch(void* const* d_in, const int* in_sizes, int n_in,
                              void* d_out, int out_size, void* d_ws, size_t ws_size,
                              hipStream_t stream) {
  (void)in_sizes; (void)n_in; (void)out_size; (void)ws_size;
  const float* x        = (const float*)d_in[0];
  const int*   mask     = (const int*)d_in[1];
  const float* Wqkv     = (const float*)d_in[2];
  const float* bqkv     = (const float*)d_in[3];
  const float* Wout     = (const float*)d_in[4];
  const float* bout     = (const float*)d_in[5];
  const float* rel_bias = (const float*)d_in[6];

  float* out  = (float*)d_out;
  float* attn = out + (size_t)B_SZ * L_SZ * D_SZ;   // attn region of d_out

  const size_t nQKV = (size_t)B_SZ * H_SZ * L_SZ * HD_SZ;  // 4,194,304 elems
  __bf16* qbf   = (__bf16*)d_ws;
  __bf16* kbf   = qbf + nQKV;
  __bf16* vbf   = kbf + nQKV;
  __bf16* ctxbf = vbf + nQKV;                        // [B*L][D] bf16

  qkv_gemm<<<dim3(N3_SZ / 128, (B_SZ * L_SZ) / 64), 256, 0, stream>>>(
      x, Wqkv, bqkv, qbf, kbf, vbf);
  attn_flash_kernel<<<dim3(L_SZ / 16, B_SZ * H_SZ), 256, 0, stream>>>(
      qbf, kbf, vbf, rel_bias, mask, attn, ctxbf);
  out_gemm<<<dim3(D_SZ / 128, (B_SZ * L_SZ) / 64), 256, 0, stream>>>(
      ctxbf, Wout, bout, out);
}
